// EncoderAttn_30382598652301
// MI455X (gfx1250) — matrix operand order repaired and checked by reading, not compiled, not run on hardware
//
#include <hip/hip_runtime.h>
#include <math.h>
#include <stdint.h>

// ---------------- problem constants (match reference) ----------------
static constexpr int B_ = 4;
static constexpr int N_ = 4096;
static constexpr int D_ = 64;
static constexpr int K_ = 16;
static constexpr int S_ = B_ * N_ * D_;   // floats per (B,N,D) tensor
static constexpr int R_ = B_ * N_ * K_;   // rows of the (B,N,K,*) tensors

typedef _Float16 f16;
typedef __attribute__((ext_vector_type(16))) _Float16 v16h;
typedef __attribute__((ext_vector_type(8)))  float    v8f;

// ---------------- WMMA fragment-order layout helpers ----------------
// A tile (16x64) stored chunked: chunk ks (k0 = ks*32) holds 512 halfs:
//   addr = ks*512 + lane*16 + e, with lane = M | (hi<<4),
//   K-in-chunk = (e&7) + ((e>=8)?16:0) + 8*hi   (ISA 7.12.2 16-bit A layout)
// B weights (64x64) stored chunked by (ks,ct): addr = (ks*4+ct)*512 + lane*16 + e,
//   K = ks*32 + (e&7) + 16*(e>>3) + 8*(lane>>4), N = ct*16 + (lane&15)
__device__ __forceinline__ int a_slot(int m, int k) {
  const int ks = k >> 5, kin = k & 31;
  const int hi = (kin >> 3) & 1;
  const int e  = (kin & 7) | ((kin & 16) >> 1);
  return ks * 512 + ((m | (hi << 4)) << 4) + e;
}
// K value touched by (lane,e) inside chunk ks of an A tile / builder loops
#define FRAG_K(ks, e, hi) ((ks) * 32 + ((e) & 7) + (((e) >> 3) << 4) + 8 * (hi))

__device__ __forceinline__ void gemm_16x64(const f16* As, const f16* Ws,
                                           v8f acc[4], int lane) {
#pragma unroll
  for (int ks = 0; ks < 2; ++ks) {
    v16h a = *(const v16h*)(As + ks * 512 + lane * 16);
#pragma unroll
    for (int ct = 0; ct < 4; ++ct) {
      v16h b = *(const v16h*)(Ws + (ks * 4 + ct) * 512 + lane * 16);
      acc[ct] = __builtin_amdgcn_wmma_f32_16x16x32_f16(
          false, a, false, b, (short)0, acc[ct], false, false);
    }
  }
}

// CDNA5 async global->LDS staging of one pre-swizzled f16 weight matrix (8 KB).
// 128 threads x 4 iters x 16 B. Tracked by ASYNCcnt.
__device__ __forceinline__ void stage_w_async(const f16* Wg, f16* Ws, int tid) {
  const uint64_t base = (uint64_t)(uintptr_t)Wg;
  const uint32_t lds  = (uint32_t)(uintptr_t)Ws;
#pragma unroll
  for (int it = 0; it < 4; ++it) {
    const uint32_t off = (uint32_t)(tid + it * 128) * 16u;
    asm volatile("global_load_async_to_lds_b128 %0, %1, %2"
                 :: "v"(lds + off), "v"(off), "s"(base)
                 : "memory");
  }
}
__device__ __forceinline__ void wait_async() {
  asm volatile("s_wait_asynccnt 0" ::: "memory");
}

// ---------------- kernel 0: zero BN stat accumulators ----------------
__global__ __launch_bounds__(256) void k_zero(float* stats) {
  stats[threadIdx.x] = 0.f;  // sumP, ssqP, sumA, ssqA (64 each)
}

// ---------------- kernel 1: convert 8 weight mats to f16 fragment order ------------
__global__ __launch_bounds__(256) void k_prep_w(
    const float* __restrict__ W0, const float* __restrict__ W1,
    const float* __restrict__ W2, const float* __restrict__ W3,
    const float* __restrict__ W4, const float* __restrict__ W5,
    const float* __restrict__ W6, const float* __restrict__ W7,
    f16* __restrict__ wf) {
  const float* Wm[8] = {W0, W1, W2, W3, W4, W5, W6, W7};
  for (int i = threadIdx.x; i < 8 * 4096; i += 256) {
    const int mat = i >> 12, s = i & 4095;
    const int chunk = s >> 9, lane = (s >> 4) & 31, e = s & 15;
    const int ks = chunk >> 2, ct = chunk & 3;
    const int k = ks * 32 + (e & 7) + ((e >> 3) << 4) + 8 * (lane >> 4);
    const int n = ct * 16 + (lane & 15);
    wf[i] = (f16)Wm[mat][k * 64 + n];
  }
}

// ---------------- kernel 2: h=x@Wpre+b ; q/k/v = h@W+b  (4 waves/WG, WMMA) ---------
__global__ __launch_bounds__(128) void k_qkv(
    const float* __restrict__ x, const f16* __restrict__ wf,
    const float* __restrict__ bpre, const float* __restrict__ bq,
    const float* __restrict__ bk, const float* __restrict__ bv,
    float* __restrict__ qo, float* __restrict__ ko, float* __restrict__ vo) {
  __shared__ __align__(32) f16 As[4][1024];
  __shared__ __align__(32) f16 Hs[4][1024];
  __shared__ __align__(32) f16 Ws[4096];
  const int tid = threadIdx.x, w = tid >> 5, lane = tid & 31;
  const int row0 = (blockIdx.x * 4 + w) * 16;
  const int m = lane & 15, hi = lane >> 4, nn = lane & 15;
  // stage x tile in A-fragment order: 2 packed v16h stores per lane
#pragma unroll
  for (int ks = 0; ks < 2; ++ks) {
    v16h a;
#pragma unroll
    for (int e = 0; e < 16; ++e) a[e] = (f16)x[(row0 + m) * 64 + FRAG_K(ks, e, hi)];
    *(v16h*)(&As[w][ks * 512 + lane * 16]) = a;
  }
  stage_w_async(wf + 0 * 4096, Ws, tid);  // Wpre
  wait_async();
  __syncthreads();
  v8f acc[4] = {};
  gemm_16x64(As[w], Ws, acc, lane);
  // h -> Hs (C layout -> A-fragment layout, cross-lane via LDS)
#pragma unroll
  for (int ct = 0; ct < 4; ++ct)
#pragma unroll
    for (int r = 0; r < 8; ++r) {
      const int mm = r + 8 * hi, c = ct * 16 + nn;
      Hs[w][a_slot(mm, c)] = (f16)(acc[ct][r] + bpre[c]);
    }
  __syncthreads();
  const float* bl[3] = {bq, bk, bv};
  float* ol[3] = {qo, ko, vo};
#pragma unroll
  for (int t = 0; t < 3; ++t) {
    stage_w_async(wf + (1 + t) * 4096, Ws, tid);  // Wq/Wk/Wv
    wait_async();
    __syncthreads();
    v8f a2[4] = {};
    gemm_16x64(Hs[w], Ws, a2, lane);
#pragma unroll
    for (int ct = 0; ct < 4; ++ct)
#pragma unroll
      for (int r = 0; r < 8; ++r) {
        const int mm = r + 8 * hi, c = ct * 16 + nn;
        ol[t][(row0 + mm) * 64 + c] = a2[ct][r] + bl[t][c];
      }
    __syncthreads();
  }
}

// ---------------- kernel 3: brute-force KNN (top-16 by squared distance) -----------
__global__ __launch_bounds__(256) void k_knn(const float* __restrict__ pc,
                                             int* __restrict__ idxb) {
  __shared__ float pcs[N_ * 3];  // 48 KB: whole-batch point cloud
  const int b = blockIdx.x >> 4;
  const int nblk = blockIdx.x & 15;
  const int tid = threadIdx.x;
  for (int i = tid; i < N_ * 3; i += 256) pcs[i] = pc[b * N_ * 3 + i];
  __syncthreads();
  const int n = nblk * 256 + tid;
  const float qx = pcs[n * 3 + 0], qy = pcs[n * 3 + 1], qz = pcs[n * 3 + 2];
  float bd[K_];
  int bi[K_];
#pragma unroll
  for (int t = 0; t < K_; ++t) { bd[t] = 3.4e38f; bi[t] = 0; }
  for (int j = 0; j < N_; ++j) {
    const float dx = pcs[j * 3 + 0] - qx;
    const float dy = pcs[j * 3 + 1] - qy;
    const float dz = pcs[j * 3 + 2] - qz;
    const float d = dx * dx + dy * dy + dz * dz;
    if (d < bd[K_ - 1]) {
      int p = K_ - 1;
      while (p > 0 && bd[p - 1] > d) { bd[p] = bd[p - 1]; bi[p] = bi[p - 1]; --p; }
      bd[p] = d; bi[p] = j;
    }
  }
#pragma unroll
  for (int t = 0; t < K_; ++t) idxb[(b * N_ + n) * K_ + t] = bi[t];
}

// ---------------- kernel 4: pos-MLP layer-1 BN stats (t1 recomputed later) ---------
__global__ __launch_bounds__(256) void k_pos_stats(
    const float* __restrict__ pc, const int* __restrict__ idxb,
    const float* __restrict__ W1, const float* __restrict__ b1,
    float* __restrict__ sum, float* __restrict__ ssq) {
  __shared__ float sA[64], sB[64];
  const int tid = threadIdx.x;
  if (tid < 64) { sA[tid] = 0.f; sB[tid] = 0.f; }
  __syncthreads();
  const int rr = blockIdx.x * 256 + tid;
  const int bn = rr >> 4;
  const int b = bn / N_, n = bn % N_;
  const int j = idxb[rr];
  const float rx = pc[(b * N_ + j) * 3 + 0] - pc[(b * N_ + n) * 3 + 0];
  const float ry = pc[(b * N_ + j) * 3 + 1] - pc[(b * N_ + n) * 3 + 1];
  const float rz = pc[(b * N_ + j) * 3 + 2] - pc[(b * N_ + n) * 3 + 2];
  for (int c = 0; c < 64; ++c) {
    const float t = rx * W1[c] + ry * W1[64 + c] + rz * W1[128 + c] + b1[c];
    atomicAdd(&sA[c], t);  // ds_add_f32
    atomicAdd(&sB[c], t * t);
  }
  __syncthreads();
  if (tid < 64) { atomicAdd(&sum[tid], sA[tid]); atomicAdd(&ssq[tid], sB[tid]); }
}

// ---------------- kernel 5: finalize BN stats ----------------
__global__ __launch_bounds__(64) void k_finstats(const float* __restrict__ sum,
                                                 const float* __restrict__ ssq,
                                                 float* __restrict__ mean,
                                                 float* __restrict__ rstd, float cnt) {
  const int c = threadIdx.x;
  const float m = sum[c] / cnt;
  const float v = ssq[c] / cnt - m * m;
  mean[c] = m;
  rstd[c] = rsqrtf(v + 1e-5f);
}

// ------- kernel 6: 4 points/WG: pos MLP (WMMA) + attn-layer-1 BN stats -------------
__global__ __launch_bounds__(128) void k_pos_pass(
    const float* __restrict__ pc, const int* __restrict__ idxb,
    const float* __restrict__ q, const float* __restrict__ kf,
    const f16* __restrict__ wf,
    const float* __restrict__ W1, const float* __restrict__ b1,
    const float* __restrict__ g1, const float* __restrict__ be1,
    const float* __restrict__ b2, const float* __restrict__ ab1,
    const float* __restrict__ meanP, const float* __restrict__ rstdP,
    float* __restrict__ posb, float* __restrict__ sumA, float* __restrict__ ssqA) {
  __shared__ __align__(32) f16 As[4][1024];
  __shared__ __align__(32) f16 Ws[4096];
  __shared__ float Pt[4][1024];
  const int tid = threadIdx.x, w = tid >> 5, lane = tid & 31;
  const int bn = blockIdx.x * 4 + w;
  const int b = bn / N_, n = bn % N_;
  const int m = lane & 15, hi = lane >> 4, nn = lane & 15;
  // this lane owns neighbor row m of its wave's tile
  const int j = idxb[bn * K_ + m];
  const float rx = pc[(b * N_ + j) * 3 + 0] - pc[(b * N_ + n) * 3 + 0];
  const float ry = pc[(b * N_ + j) * 3 + 1] - pc[(b * N_ + n) * 3 + 1];
  const float rz = pc[(b * N_ + j) * 3 + 2] - pc[(b * N_ + n) * 3 + 2];
  // t1 = rel@W1+b1 (3 FMAs/elem, recomputed), BN + ReLU -> A-fragment tile
#pragma unroll
  for (int ks = 0; ks < 2; ++ks) {
    v16h a;
#pragma unroll
    for (int e = 0; e < 16; ++e) {
      const int k = FRAG_K(ks, e, hi);
      const float t = rx * W1[k] + ry * W1[64 + k] + rz * W1[128 + k] + b1[k];
      const float xn = (t - meanP[k]) * rstdP[k] * g1[k] + be1[k];
      a[e] = (f16)fmaxf(xn, 0.f);
    }
    *(v16h*)(&As[w][ks * 512 + lane * 16]) = a;
  }
  stage_w_async(wf + 4 * 4096, Ws, tid);  // pos_W2
  wait_async();
  __syncthreads();
  v8f acc[4] = {};
  gemm_16x64(As[w], Ws, acc, lane);
  // pos -> global + LDS f32 tile (cross-lane handoff for the a_in builder)
#pragma unroll
  for (int ct = 0; ct < 4; ++ct)
#pragma unroll
    for (int r = 0; r < 8; ++r) {
      const int mm = r + 8 * hi, c = ct * 16 + nn;
      const float pv = acc[ct][r] + b2[c];
      posb[(bn * K_ + mm) * 64 + c] = pv;
      Pt[w][mm * 64 + c] = pv;
    }
  __syncthreads();  // all waves done with Ws + Pt visible
  // a_in = kf[idx] - q + pos -> A-fragment tile
#pragma unroll
  for (int ks = 0; ks < 2; ++ks) {
    v16h a;
#pragma unroll
    for (int e = 0; e < 16; ++e) {
      const int k = FRAG_K(ks, e, hi);
      a[e] = (f16)(kf[(b * N_ + j) * 64 + k] - q[bn * 64 + k] + Pt[w][m * 64 + k]);
    }
    *(v16h*)(&As[w][ks * 512 + lane * 16]) = a;
  }
  stage_w_async(wf + 5 * 4096, Ws, tid);  // att_W1
  wait_async();
  __syncthreads();
  v8f a2[4] = {};
  gemm_16x64(As[w], Ws, a2, lane);
  // attn-layer-1 BN statistics (t2 recomputed in k_attn_out)
#pragma unroll
  for (int ct = 0; ct < 4; ++ct) {
    float s = 0.f, s2 = 0.f;
#pragma unroll
    for (int r = 0; r < 8; ++r) {
      const float t = a2[ct][r] + ab1[ct * 16 + nn];
      s += t; s2 += t * t;
    }
    atomicAdd(&sumA[ct * 16 + nn], s);
    atomicAdd(&ssqA[ct * 16 + nn], s2);
  }
}

// ------- kernel 7: 4 points/WG: attn MLP (WMMA x2) + softmax + aggregation ---------
__global__ __launch_bounds__(128) void k_attn_out(
    const int* __restrict__ idxb, const float* __restrict__ q,
    const float* __restrict__ kf, const float* __restrict__ vf,
    const float* __restrict__ posb, const f16* __restrict__ wf,
    const float* __restrict__ ab1, const float* __restrict__ ag,
    const float* __restrict__ abe, const float* __restrict__ ab2,
    const float* __restrict__ meanA, const float* __restrict__ rstdA,
    float* __restrict__ outp) {
  __shared__ __align__(32) f16 As[4][1024];
  __shared__ __align__(32) f16 Ws[4096];
  __shared__ float Ps[4][1024];  // pos tiles (f32) for aggregation
  __shared__ float Ls[4][1024];  // attention logits
  const int tid = threadIdx.x, w = tid >> 5, lane = tid & 31;
  const int bn = blockIdx.x * 4 + w;
  const int b = bn / N_;
  const int m = lane & 15, hi = lane >> 4, nn = lane & 15;
  const int j = idxb[bn * K_ + m];
#pragma unroll
  for (int ks = 0; ks < 2; ++ks) {
    v16h a;
#pragma unroll
    for (int e = 0; e < 16; ++e) {
      const int k = FRAG_K(ks, e, hi);
      const float pv = posb[(bn * K_ + m) * 64 + k];
      Ps[w][m * 64 + k] = pv;  // lanes hi=0/1 cover disjoint k halves
      a[e] = (f16)(kf[(b * N_ + j) * 64 + k] - q[bn * 64 + k] + pv);
    }
    *(v16h*)(&As[w][ks * 512 + lane * 16]) = a;
  }
  stage_w_async(wf + 5 * 4096, Ws, tid);  // att_W1
  wait_async();
  __syncthreads();
  v8f acc[4] = {};
  gemm_16x64(As[w], Ws, acc, lane);
  __syncthreads();  // all waves done with Ws before restage
  // BN + ReLU -> A-fragment tile (cross-lane)
#pragma unroll
  for (int ct = 0; ct < 4; ++ct)
#pragma unroll
    for (int r = 0; r < 8; ++r) {
      const int mm = r + 8 * hi, c = ct * 16 + nn;
      const float t = acc[ct][r] + ab1[c];
      const float xn = (t - meanA[c]) * rstdA[c] * ag[c] + abe[c];
      As[w][a_slot(mm, c)] = (f16)fmaxf(xn, 0.f);
    }
  stage_w_async(wf + 6 * 4096, Ws, tid);  // att_W2
  wait_async();
  __syncthreads();
  v8f a2[4] = {};
  gemm_16x64(As[w], Ws, a2, lane);
#pragma unroll
  for (int ct = 0; ct < 4; ++ct)
#pragma unroll
    for (int r = 0; r < 8; ++r) {
      const int mm = r + 8 * hi, c = ct * 16 + nn;
      Ls[w][mm * 64 + c] = a2[ct][r] + ab2[c];
    }
  __syncthreads();
  // per-column softmax over 16 neighbor rows + weighted sum of (v + pos)
#pragma unroll
  for (int half = 0; half < 2; ++half) {
    const int c = lane + 32 * half;
    float mx = -3.4e38f;
#pragma unroll
    for (int kk = 0; kk < K_; ++kk) mx = fmaxf(mx, Ls[w][kk * 64 + c]);
    float e[K_];
    float s = 0.f;
#pragma unroll
    for (int kk = 0; kk < K_; ++kk) { e[kk] = __expf(Ls[w][kk * 64 + c] - mx); s += e[kk]; }
    const float inv = 1.f / s;
    float o = 0.f;
#pragma unroll
    for (int kk = 0; kk < K_; ++kk) {
      const int jj = idxb[bn * K_ + kk];
      o += e[kk] * inv * (vf[(b * N_ + jj) * 64 + c] + Ps[w][kk * 64 + c]);
    }
    outp[bn * 64 + c] = o;
  }
}

// ---------------- kernel 8: out@Wpost + bpost + residual; pc passthrough -----------
__global__ __launch_bounds__(128) void k_post(
    const float* __restrict__ outp, const float* __restrict__ x,
    const float* __restrict__ pc, const f16* __restrict__ wf,
    const float* __restrict__ bpost, float* __restrict__ dout) {
  __shared__ __align__(32) f16 As[4][1024];
  __shared__ __align__(32) f16 Ws[4096];
  const int tid = threadIdx.x, w = tid >> 5, lane = tid & 31;
  const int row0 = (blockIdx.x * 4 + w) * 16;
  const int m = lane & 15, hi = lane >> 4, nn = lane & 15;
#pragma unroll
  for (int ks = 0; ks < 2; ++ks) {
    v16h a;
#pragma unroll
    for (int e = 0; e < 16; ++e) a[e] = (f16)outp[(row0 + m) * 64 + FRAG_K(ks, e, hi)];
    *(v16h*)(&As[w][ks * 512 + lane * 16]) = a;
  }
  stage_w_async(wf + 7 * 4096, Ws, tid);  // Wpost
  wait_async();
  __syncthreads();
  v8f acc[4] = {};
  gemm_16x64(As[w], Ws, acc, lane);
#pragma unroll
  for (int ct = 0; ct < 4; ++ct)
#pragma unroll
    for (int r = 0; r < 8; ++r) {
      const int mm = r + 8 * hi, c = ct * 16 + nn;
      dout[(row0 + mm) * 64 + c] = acc[ct][r] + bpost[c] + x[(row0 + mm) * 64 + c];
    }
  const int rowb = blockIdx.x * 64;
  for (int i = tid; i < 64 * 3; i += 128)
    dout[B_ * N_ * 64 + rowb * 3 + i] = pc[rowb * 3 + i];
}

// ---------------- launch ----------------
extern "C" void kernel_launch(void* const* d_in, const int* in_sizes, int n_in,
                              void* d_out, int out_size, void* d_ws, size_t ws_size,
                              hipStream_t stream) {
  const float* x    = (const float*)d_in[0];
  const float* pc   = (const float*)d_in[1];
  const float* Wpre = (const float*)d_in[2];  const float* bpre = (const float*)d_in[3];
  const float* Wq   = (const float*)d_in[4];  const float* bq   = (const float*)d_in[5];
  const float* Wk   = (const float*)d_in[6];  const float* bk   = (const float*)d_in[7];
  const float* Wv   = (const float*)d_in[8];  const float* bv   = (const float*)d_in[9];
  const float* pW1  = (const float*)d_in[10]; const float* pb1  = (const float*)d_in[11];
  const float* pg   = (const float*)d_in[12]; const float* pbe  = (const float*)d_in[13];
  const float* pW2  = (const float*)d_in[14]; const float* pb2  = (const float*)d_in[15];
  const float* aW1  = (const float*)d_in[16]; const float* ab1  = (const float*)d_in[17];
  const float* ag   = (const float*)d_in[18]; const float* abe  = (const float*)d_in[19];
  const float* aW2  = (const float*)d_in[20]; const float* ab2  = (const float*)d_in[21];
  const float* Wpost= (const float*)d_in[22]; const float* bpost= (const float*)d_in[23];
  (void)in_sizes; (void)n_in; (void)out_size; (void)ws_size;

  float* ws   = (float*)d_ws;
  float* qb   = ws;                     // (B,N,64)
  float* kb   = ws + S_;                // (B,N,64)
  float* vb   = ws + 2 * S_;            // (B,N,64)
  float* posb = ws + 3 * S_;            // (B,N,K,64)
  float* outp = ws + 3 * S_ + R_ * 64;  // (B,N,64)
  int*   idxb = (int*)(ws + 4 * S_ + R_ * 64);   // (B,N,K) int
  float* stats= ws + 4 * S_ + R_ * 64 + R_;      // 512 floats
  f16*   wsW  = (f16*)(stats + 512);             // 8 x 4096 f16 fragment-order weights
  float* sumP = stats;        float* ssqP = stats + 64;
  float* sumA = stats + 128;  float* ssqA = stats + 192;
  float* meanP= stats + 256;  float* rstdP= stats + 320;
  float* meanA= stats + 384;  float* rstdA= stats + 448;

  k_zero<<<1, 256, 0, stream>>>(stats);
  k_prep_w<<<1, 256, 0, stream>>>(Wpre, Wq, Wk, Wv, pW2, aW1, aW2, Wpost, wsW);
  k_qkv<<<B_ * N_ / 64, 128, 0, stream>>>(x, wsW, bpre, bq, bk, bv, qb, kb, vb);
  k_knn<<<B_ * (N_ / 256), 256, 0, stream>>>(pc, idxb);
  k_pos_stats<<<R_ / 256, 256, 0, stream>>>(pc, idxb, pW1, pb1, sumP, ssqP);
  k_finstats<<<1, 64, 0, stream>>>(sumP, ssqP, meanP, rstdP, (float)R_);
  k_pos_pass<<<B_ * N_ / 4, 128, 0, stream>>>(pc, idxb, qb, kb, wsW, pW1, pb1, pg, pbe,
                                              pb2, ab1, meanP, rstdP, posb, sumA, ssqA);
  k_finstats<<<1, 64, 0, stream>>>(sumA, ssqA, meanA, rstdA, (float)R_);
  k_attn_out<<<B_ * N_ / 4, 128, 0, stream>>>(idxb, qb, kb, vb, posb, wsW, ab1, ag, abe,
                                              ab2, meanA, rstdA, outp);
  k_post<<<B_ * N_ / 64, 128, 0, stream>>>(outp, x, pc, wsW, bpost, (float*)d_out);
}
